// gat_cell_13219909337614
// MI455X (gfx1250) — matrix operand order repaired and checked by reading, not compiled, not run on hardware
//
#include <hip/hip_runtime.h>

// ---------------------------------------------------------------------------
// GAT forward, fused, for MI455X (gfx1250, wave32, WMMA bf16 16x16x32)
// B=8, N=2048, H=128
// ---------------------------------------------------------------------------

constexpr int GB = 8;
constexpr int GN = 2048;
constexpr int GH = 128;
constexpr float LRELU_ALPHA = 0.2f;

typedef __attribute__((ext_vector_type(16))) __bf16        v16bf;
typedef __attribute__((ext_vector_type(8)))  float         v8f;
typedef __attribute__((ext_vector_type(4)))  float         v4f;
typedef __attribute__((ext_vector_type(4)))  unsigned int  v4u;

union Frag {
    v16bf          v;
    unsigned int   d[8];
    v4u            q[2];
};

// Pack two f32 -> bf16x2 dword: round-half-up then one v_perm_b32.
__device__ __forceinline__ unsigned int pack2bf(float lo, float hi) {
    union { float f; unsigned int u; } a, b;
    a.f = lo; b.f = hi;
    unsigned int x = a.u + 0x8000u;
    unsigned int y = b.u + 0x8000u;
    // result bytes: [1:0] = x bytes [3:2], [3:2] = y bytes [3:2]
    return __builtin_amdgcn_perm(y, x, 0x07060302u);
}

__device__ __forceinline__ unsigned short f2bf(float f) {
    union { float f; unsigned int u; } x; x.f = f;
    return (unsigned short)((x.u + 0x8000u) >> 16);
}

// ---------------------------------------------------------------------------
// Kernel 1: Wh = inp @ W (WMMA bf16), store Wh^T as bf16, compute Wh1/Wh2.
//  - 8 waves per block, one 16-row tile of (B*N) per wave.
//  - W^T staged in LDS as bf16; all 8 B-fragments of a K-chunk preloaded so
//    ds_load_b128s issue back-to-back ahead of the WMMA burst.
// ---------------------------------------------------------------------------
__global__ __launch_bounds__(256) void gat_wh_kernel(
    const float* __restrict__ inp, const float* __restrict__ W,
    const float* __restrict__ avec,
    unsigned short* __restrict__ whT,
    float* __restrict__ wh1, float* __restrict__ wh2)
{
    __shared__ unsigned short sWt[GH][136];   // W transposed, bf16, padded rows

    const int t = threadIdx.x;
    for (int idx = t; idx < GH * GH; idx += 256) {
        int k = idx >> 7, n = idx & 127;
        sWt[n][k] = f2bf(W[idx]);             // W is [k][n] row-major
    }
    __syncthreads();

    const int lane  = t & 31;
    const int wv    = t >> 5;
    const int tile  = blockIdx.x * 8 + wv;    // 16-row tile over B*N (1024 tiles)
    const int b     = tile >> 7;              // 128 tiles per batch
    const int row0  = (tile & 127) << 4;      // row within batch
    const int mrow  = lane & 15;              // tile-row / tile-col for this lane
    const int khalf = lane >> 4;

    const float* inpRow = inp + ((size_t)b * GN + row0 + mrow) * GH;

    v8f acc[8] = {};

#pragma unroll
    for (int kc = 0; kc < 4; ++kc) {
        // Preload all 8 B fragments of this K-chunk (16 x ds_load_b128).
        Frag fbs[8];
#pragma unroll
        for (int nt = 0; nt < 8; ++nt) {
            const v4u* q4 = (const v4u*)&sWt[nt * 16 + mrow][kc * 32 + khalf * 16];
            fbs[nt].q[0] = q4[0];
            fbs[nt].q[1] = q4[1];
        }

        // A fragment: rows of inp tile, one contiguous 16-float run per lane.
        Frag fa;
        const float* p0 = inpRow + kc * 32 + khalf * 16;
        v4f x0 = *(const v4f*)(p0);
        v4f x1 = *(const v4f*)(p0 + 4);
        v4f x2 = *(const v4f*)(p0 + 8);
        v4f x3 = *(const v4f*)(p0 + 12);
        fa.d[0] = pack2bf(x0[0], x0[1]);
        fa.d[1] = pack2bf(x0[2], x0[3]);
        fa.d[2] = pack2bf(x1[0], x1[1]);
        fa.d[3] = pack2bf(x1[2], x1[3]);
        fa.d[4] = pack2bf(x2[0], x2[1]);
        fa.d[5] = pack2bf(x2[2], x2[3]);
        fa.d[6] = pack2bf(x3[0], x3[1]);
        fa.d[7] = pack2bf(x3[2], x3[3]);

#pragma unroll
        for (int nt = 0; nt < 8; ++nt) {
            acc[nt] = __builtin_amdgcn_wmma_f32_16x16x32_bf16(
                false, fa.v, false, fbs[nt].v, (short)0, acc[nt], false, false);
        }
    }

    const int rsel = khalf << 3;              // upper lane-half holds rows +8

    // Store Wh^T (bf16): the 8 accumulator rows are contiguous in N -> b128.
#pragma unroll
    for (int nt = 0; nt < 8; ++nt) {
        v4u pk;
#pragma unroll
        for (int j = 0; j < 4; ++j)
            pk[j] = pack2bf(acc[nt][2 * j], acc[nt][2 * j + 1]);
        *(v4u*)(whT + ((size_t)b * GH + nt * 16 + mrow) * GN + row0 + rsel) = pk;
    }

    // Wh1/Wh2 per-row reductions.
    float a1v[8], a2v[8];
#pragma unroll
    for (int nt = 0; nt < 8; ++nt) {
        a1v[nt] = avec[nt * 16 + mrow];
        a2v[nt] = avec[GH + nt * 16 + mrow];
    }
#pragma unroll
    for (int r = 0; r < 8; ++r) {
        const int grow = row0 + r + rsel;
        float s1 = 0.f, s2 = 0.f;
#pragma unroll
        for (int nt = 0; nt < 8; ++nt) {
            float vh = acc[nt][r];
            s1 = fmaf(vh, a1v[nt], s1);
            s2 = fmaf(vh, a2v[nt], s2);
        }
#pragma unroll
        for (int off = 8; off; off >>= 1) {
            s1 += __shfl_xor(s1, off, 16);
            s2 += __shfl_xor(s2, off, 16);
        }
        if (mrow == 0) {
            wh1[(size_t)b * GN + grow] = s1;
            wh2[(size_t)b * GN + grow] = s2;
        }
    }
}

// ---------------------------------------------------------------------------
// Kernel 2: fused masked-softmax + attention @ Wh + ELU.
// One block per (b, 16-row tile). 3 passes over the A row-tile (L2 resident):
//   1) row max, 2) row sum-exp, 3) build p=exp(e-m) bf16 A-frags, WMMA against
//      Wh^T bf16 B-frags, merge wave partials via ds_add_f32, finalize.
// In pass 3 the 8 Wh^T B-fragments are preloaded (global_load_b128 burst) and
// the exp/leaky-relu A-fragment VALU work hides their latency.
// ---------------------------------------------------------------------------
__global__ __launch_bounds__(256) void gat_attn_kernel(
    const float* __restrict__ Adj, const unsigned short* __restrict__ whT,
    const float* __restrict__ wh1, const float* __restrict__ wh2,
    float* __restrict__ out)
{
    __shared__ float sM[16];
    __shared__ float sL[16];
    __shared__ float sAcc[16][GH];

    const int t    = threadIdx.x;
    const int tile = blockIdx.x;
    const int b    = tile >> 7;
    const int row0 = (tile & 127) << 4;

    for (int idx = t; idx < 16 * GH; idx += 256) ((float*)sAcc)[idx] = 0.f;

    const int    r     = t >> 4;
    const int    c     = t & 15;
    const int    growr = row0 + r;
    const float* Arow  = Adj + ((size_t)b * GN + growr) * GN;
    const float* wh2b  = wh2 + (size_t)b * GN;
    const float  wh1r  = wh1[(size_t)b * GN + growr];

    // ---- pass 1: per-row max of masked leaky-relu scores ----
    float m = -3.0e38f;
    for (int jj = c * 4; jj < GN; jj += 64) {
        __builtin_prefetch(Arow + jj + 256, 0, 1);
        v4f av = *(const v4f*)(Arow + jj);
        v4f w2 = *(const v4f*)(wh2b + jj);
#pragma unroll
        for (int q = 0; q < 4; ++q) {
            float aa = av[q] + ((jj + q) == growr ? 1.f : 0.f);
            if (aa > 0.f) {
                float e = (wh1r + w2[q]) * aa;
                e = e >= 0.f ? e : LRELU_ALPHA * e;
                m = fmaxf(m, e);
            }
        }
    }
#pragma unroll
    for (int off = 8; off; off >>= 1) m = fmaxf(m, __shfl_xor(m, off, 16));
    if (c == 0) sM[r] = m;
    __syncthreads();
    m = sM[r];

    // ---- pass 2: per-row softmax denominator ----
    float l = 0.f;
    for (int jj = c * 4; jj < GN; jj += 64) {
        v4f av = *(const v4f*)(Arow + jj);
        v4f w2 = *(const v4f*)(wh2b + jj);
#pragma unroll
        for (int q = 0; q < 4; ++q) {
            float aa = av[q] + ((jj + q) == growr ? 1.f : 0.f);
            if (aa > 0.f) {
                float e = (wh1r + w2[q]) * aa;
                e = e >= 0.f ? e : LRELU_ALPHA * e;
                l += __expf(e - m);
            }
        }
    }
#pragma unroll
    for (int off = 8; off; off >>= 1) l += __shfl_xor(l, off, 16);
    if (c == 0) sL[r] = l;
    __syncthreads();

    // ---- pass 3: p = exp(e - m) (bf16) ; acc += p @ Wh via WMMA ----
    const int    lane  = t & 31;
    const int    wv    = t >> 5;
    const int    mrow  = lane & 15;
    const int    khalf = lane >> 4;
    const int    grf   = row0 + mrow;
    const float  mr    = sM[mrow];
    const float  wh1f  = wh1[(size_t)b * GN + grf];
    const float* ArowF = Adj + ((size_t)b * GN + grf) * GN;
    const unsigned short* wbase = whT + (size_t)b * GH * GN;

    v8f acc[8] = {};

    for (int cc = 0; cc < 8; ++cc) {
        const int jbase = (wv * 8 + cc) * 32;   // this wave's 32-wide K chunk

        // Preload the 8 Wh^T B-fragments (16 x global_load_b128, back-to-back).
        Frag fbs[8];
#pragma unroll
        for (int nt = 0; nt < 8; ++nt) {
            const v4u* q4 = (const v4u*)(wbase + ((size_t)(nt * 16 + mrow)) * GN
                                         + jbase + khalf * 16);
            fbs[nt].q[0] = q4[0];
            fbs[nt].q[1] = q4[1];
        }

        // Build the p = exp(e - m) A-fragment (hides the B-frag load latency).
        Frag fp;
#pragma unroll
        for (int hh = 0; hh < 2; ++hh) {
            const int jb = jbase + khalf * 16 + hh * 8;
            v4f a0 = *(const v4f*)(ArowF + jb);
            v4f a1 = *(const v4f*)(ArowF + jb + 4);
            v4f w0 = *(const v4f*)(wh2b + jb);
            v4f w1 = *(const v4f*)(wh2b + jb + 4);
            float p[8];
#pragma unroll
            for (int q = 0; q < 8; ++q) {
                float av = (q < 4) ? a0[q] : a1[q - 4];
                float wq = (q < 4) ? w0[q] : w1[q - 4];
                float aa = av + ((jb + q) == grf ? 1.f : 0.f);
                float pv = 0.f;
                if (aa > 0.f) {
                    float e = (wh1f + wq) * aa;
                    e = e >= 0.f ? e : LRELU_ALPHA * e;
                    pv = __expf(e - mr);
                }
                p[q] = pv;
            }
#pragma unroll
            for (int j = 0; j < 4; ++j)
                fp.d[hh * 4 + j] = pack2bf(p[2 * j], p[2 * j + 1]);
        }

#pragma unroll
        for (int nt = 0; nt < 8; ++nt) {
            acc[nt] = __builtin_amdgcn_wmma_f32_16x16x32_bf16(
                false, fp.v, false, fbs[nt].v, (short)0, acc[nt], false, false);
        }
    }

    // Merge wave partials into LDS with ds_add_f32 atomics.
    const int rsel = khalf << 3;
#pragma unroll
    for (int nt = 0; nt < 8; ++nt) {
#pragma unroll
        for (int rr = 0; rr < 8; ++rr) {
            atomicAdd(&sAcc[rr + rsel][nt * 16 + mrow], acc[nt][rr]);
        }
    }
    __syncthreads();

    // ---- finalize: divide by softmax denom, ELU, store ----
    for (int idx = t; idx < 16 * GH; idx += 256) {
        int rr  = idx >> 7;
        int col = idx & 127;
        float h = sAcc[rr][col] / sL[rr];
        h = h > 0.f ? h : (__expf(h) - 1.f);
        out[((size_t)b * GN + row0 + rr) * GH + col] = h;
    }
}

// ---------------------------------------------------------------------------
extern "C" void kernel_launch(void* const* d_in, const int* in_sizes, int n_in,
                              void* d_out, int out_size, void* d_ws, size_t ws_size,
                              hipStream_t stream) {
    const float* inp  = (const float*)d_in[0];   // [8,2048,128]
    const float* Adj  = (const float*)d_in[1];   // [8,2048,2048]
    const float* W    = (const float*)d_in[2];   // [128,128]
    const float* avec = (const float*)d_in[3];   // [256,1]
    float*       out  = (float*)d_out;           // [8,2048,128]

    // Workspace: Wh^T bf16 [B][H][N] (4 MB) + Wh1 (64 KB) + Wh2 (64 KB)
    unsigned short* whT = (unsigned short*)d_ws;
    const size_t whT_elems = (size_t)GB * GH * GN;
    float* wh1 = (float*)((char*)d_ws + whT_elems * sizeof(unsigned short));
    float* wh2 = wh1 + (size_t)GB * GN;

    gat_wh_kernel<<<dim3(128), dim3(256), 0, stream>>>(inp, W, avec, whT, wh1, wh2);
    gat_attn_kernel<<<dim3(GB * GN / 16), dim3(256), 0, stream>>>(Adj, whT, wh1, wh2, out);
}
